// SNNDirectionDecoder_48369921688010
// MI455X (gfx1250) — compile-verified, matrix operand order checked by reading, not run on hardware
//
#include <hip/hip_runtime.h>
#include <hip/hip_bf16.h>

typedef __attribute__((ext_vector_type(16))) __bf16        v16bf;
typedef __attribute__((ext_vector_type(8)))  float         v8f;
typedef __attribute__((ext_vector_type(4)))  unsigned int  v4u_t;
typedef __attribute__((ext_vector_type(8)))  int           v8i_t;
typedef __attribute__((ext_vector_type(4)))  int           v4i_t;

#define T_STEPS 100
#define IN_DIM  512
#define HID     1024
#define BATCH   256
#define NCLS    8

#define BM 16                 // batch rows per workgroup (WMMA M)
#define HN 64                 // hidden cols per workgroup (16 per wave)
#define KSTEPS (IN_DIM / 32)  // 16 WMMA k-steps of 32
#define APITCH (IN_DIM + 8)   // padded LDS pitch (ushorts): pitch 1040B -> conflict-free frags

#if defined(__gfx1250__) && __has_builtin(__builtin_amdgcn_tensor_load_to_lds)
#define USE_TDM 1
#endif
#if __has_include(<hip/amd_detail/amd_gfx1250_TDM.h>)
#define TDM_6ARG 1
#endif

__device__ __forceinline__ unsigned bf16_rne(float f) {
  unsigned u = __float_as_uint(f);
  return ((u + 0x7FFFu + ((u >> 16) & 1u)) >> 16) & 0xFFFFu;
}
__device__ __forceinline__ unsigned pack2(float a, float b) {
  return bf16_rne(a) | (bf16_rne(b) << 16);
}

union Frag { v16bf v; uint4 q[2]; };

#ifdef USE_TDM
// ---- Tensor DMA descriptor (D#) per CDNA5 ISA ----
// Group 0: count=1 | lds_addr | global_addr[56:0] | type=2
__device__ __forceinline__ v4u_t tdm_g0(unsigned lds_addr, unsigned long long gaddr) {
  v4u_t g;
  g[0] = 1u;                                   // count=1, is_restore=0, gather off
  g[1] = lds_addr;                             // bits 63:32
  g[2] = (unsigned)gaddr;                      // global_addr[31:0]
  g[3] = (unsigned)((gaddr >> 32) & 0x01FFFFFFull) | (2u << 30); // addr[56:32], type=2
  return g;
}
// Group 1: 2D tile 512x16 of 2-byte elems, row stride T*I elems,
// LDS pad: 4 dwords (16B) after every 256 dwords (1024B) -> pitch 1040B == APITCH.
__device__ __forceinline__ v8i_t tdm_g1() {
  v8i_t g;
  g[0] = (int)((1u << 16)                      // data_size = 2 bytes
             | (1u << 20)                      // pad_enable
             | (7u << 22)                      // pad_interval: 256 dwords
             | (3u << 25));                    // pad_amount: 4 dwords
  g[1] = (int)(((unsigned)IN_DIM & 0xFFFFu) << 16);   // tensor_dim0[15:0]
  g[2] = (int)((((unsigned)IN_DIM >> 16) & 0xFFFFu)   // tensor_dim0[31:16]
             | ((unsigned)BM << 16));                 // tensor_dim1[15:0] = 16
  g[3] = (int)((unsigned)IN_DIM << 16);               // tile_dim0 = 512
  g[4] = (int)(unsigned)BM;                           // tile_dim1 = 16, tile_dim2 = 0
  g[5] = (int)(unsigned)(T_STEPS * IN_DIM);           // tensor_dim0_stride (elems)
  g[6] = 0;
  g[7] = 0;
  return g;
}
__device__ __forceinline__ void tdm_issue(unsigned lds_addr, unsigned long long gaddr,
                                          v8i_t g1) {
  const v4i_t z4 = {};
#ifdef TDM_6ARG
  const v8i_t z8 = {};
  __builtin_amdgcn_tensor_load_to_lds(tdm_g0(lds_addr, gaddr), g1, z4, z4, z8, 0);
#else
  __builtin_amdgcn_tensor_load_to_lds(tdm_g0(lds_addr, gaddr), g1, z4, z4, 0);
#endif
}
#endif // USE_TDM

// ---- Pre-pass: x f32 -> bf16 (RNE), done once instead of 100x in the hot loop.
__global__ __launch_bounds__(256)
void x_to_bf16_kernel(const float* __restrict__ x, unsigned short* __restrict__ xb)
{
  const size_t i8 = ((size_t)blockIdx.x * 256 + threadIdx.x) * 8;
  const float4 a = *(const float4*)(x + i8);
  const float4 b = *(const float4*)(x + i8 + 4);
  *(uint4*)(xb + i8) = make_uint4(pack2(a.x, a.y), pack2(a.z, a.w),
                                  pack2(b.x, b.y), pack2(b.z, b.w));
}

__global__ __launch_bounds__(128)
void snn_fused_kernel(const unsigned short* __restrict__ xb,  // x in bf16 [B,T,I]
                      const float* __restrict__ W1,
                      const float* __restrict__ b1,
                      float* __restrict__ spk_sum)
{
  __shared__ __align__(16) unsigned short As[2][BM][APITCH];

  const int tid   = threadIdx.x;
  const int lane  = tid & 31;
  const int wv    = tid >> 5;       // 0..3 : wave's 16-wide h sub-tile
  const int htile = blockIdx.x;     // 0..15
  const int btile = blockIdx.y;     // 0..15

  const int n  = lane & 15;         // frag column (B/C) / frag row (A)
  const int hi = lane >> 4;         // lane-half selector in frag layout
  const int hg = htile * HN + wv * 16 + n;

  // ---- Preload B fragments (W1^T tile) into registers, f32 -> bf16 (once).
  Frag Bf[KSTEPS];
  {
    const float* wrow = W1 + (size_t)hg * IN_DIM;
    #pragma unroll
    for (int ks = 0; ks < KSTEPS; ++ks) {
      const int k0 = ks * 32;
      const float4 f0 = *(const float4*)(wrow + k0 + hi * 8);
      const float4 f1 = *(const float4*)(wrow + k0 + hi * 8 + 4);
      const float4 f2 = *(const float4*)(wrow + k0 + 16 + hi * 8);
      const float4 f3 = *(const float4*)(wrow + k0 + 16 + hi * 8 + 4);
      Bf[ks].q[0] = make_uint4(pack2(f0.x, f0.y), pack2(f0.z, f0.w),
                               pack2(f1.x, f1.y), pack2(f1.z, f1.w));
      Bf[ks].q[1] = make_uint4(pack2(f2.x, f2.y), pack2(f2.z, f2.w),
                               pack2(f3.x, f3.y), pack2(f3.z, f3.w));
    }
  }

  const float b1h = b1[hg];

  v8f mem  = {};   // membrane potential: 8 batch cells (fixed h) per lane
  v8f ssum = {};   // spike-count accumulator

#ifdef USE_TDM
  // ---- A-tile staging via Tensor Data Mover, double-buffered in LDS.
  const v8i_t g1 = tdm_g1();
  const unsigned long long xbase =
      (unsigned long long)(size_t)(xb + (size_t)(btile * BM) * T_STEPS * IN_DIM);
  const unsigned ldsA[2] = { (unsigned)(size_t)&As[0][0][0],
                             (unsigned)(size_t)&As[1][0][0] };
  int p = 0;
  if (wv == 0) {
    tdm_issue(ldsA[0], xbase, g1);                       // tile t=0
    __builtin_amdgcn_s_wait_tensorcnt(0);
  }
  __syncthreads();

  for (int t = 0; t < T_STEPS; ++t) {
    if (t + 1 < T_STEPS && wv == 0)                      // prefetch t+1 into back buffer
      tdm_issue(ldsA[p ^ 1], xbase + (unsigned long long)(t + 1) * IN_DIM * 2, g1);
#else
  int p = 0;
  for (int t = 0; t < T_STEPS; ++t) {
    { // fallback: cooperative bf16 copy global -> LDS (single buffer)
      const int arow = tid >> 3, i0 = (tid & 7) * 64;
      const unsigned short* xr =
          xb + ((size_t)(btile * BM + arow) * T_STEPS + t) * IN_DIM + i0;
      __syncthreads();
      #pragma unroll
      for (int q = 0; q < 8; ++q)
        *(uint4*)&As[0][arow][i0 + q * 8] = ((const uint4*)xr)[q];
      __syncthreads();
    }
#endif

    // ---- GEMM tile: software-pipelined A-frag loads (ping-pong) + dual
    // accumulator chains so ds latency and XDL latency overlap.
    const unsigned short* arow_p = &As[p][n][hi * 8];
    v8f acc0 = {};
    v8f acc1 = {};
    Frag Afa, Afb;
    Afa.q[0] = *(const uint4*)(arow_p + 0);
    Afa.q[1] = *(const uint4*)(arow_p + 16);
    #pragma unroll
    for (int ks = 0; ks < KSTEPS; ++ks) {
      Frag& cur = (ks & 1) ? Afb : Afa;
      Frag& nxt = (ks & 1) ? Afa : Afb;
      if (ks + 1 < KSTEPS) {
        const int k0 = (ks + 1) * 32;
        nxt.q[0] = *(const uint4*)(arow_p + k0);
        nxt.q[1] = *(const uint4*)(arow_p + k0 + 16);
      }
      if (ks & 1)
        acc1 = __builtin_amdgcn_wmma_f32_16x16x32_bf16(
                   false, cur.v, false, Bf[ks].v, (short)0, acc1, false, false);
      else
        acc0 = __builtin_amdgcn_wmma_f32_16x16x32_bf16(
                   false, cur.v, false, Bf[ks].v, (short)0, acc0, false, false);
    }
    const v8f acc = acc0 + acc1;

    // LIF (snntorch Leaky, reset-by-subtraction), THRESH=1, BETA=0.9.
    #pragma unroll
    for (int r = 0; r < 8; ++r) {
      const float cur   = acc[r] + b1h;
      const float reset = (mem[r] > 1.0f) ? 1.0f : 0.0f;
      mem[r] = 0.9f * mem[r] + cur - reset;
      ssum[r] += (mem[r] > 1.0f) ? 1.0f : 0.0f;
    }

#ifdef USE_TDM
    if (wv == 0) __builtin_amdgcn_s_wait_tensorcnt(0);   // t+1 tile landed
    __syncthreads();                                     // everyone done with buf p
    p ^= 1;
#endif
  }

  // C-frag layout: lane holds N=hg fixed; VGPR r -> batch row r + 8*hi.
  #pragma unroll
  for (int r = 0; r < 8; ++r) {
    const int bg = btile * BM + r + 8 * hi;
    spk_sum[(size_t)bg * HID + hg] = ssum[r];
  }
}

__global__ __launch_bounds__(256)
void snn_head_kernel(const float* __restrict__ spk_sum,
                     const float* __restrict__ W2,
                     const float* __restrict__ b2,
                     float* __restrict__ out)
{
  __shared__ float red[NCLS][256];
  const int tid = threadIdx.x;
  const int b   = blockIdx.x;

  float acc[NCLS];
  #pragma unroll
  for (int c = 0; c < NCLS; ++c) acc[c] = 0.0f;

  for (int h = tid; h < HID; h += 256) {
    const float s = spk_sum[(size_t)b * HID + h];
    #pragma unroll
    for (int c = 0; c < NCLS; ++c) acc[c] += s * W2[c * HID + h];
  }
  #pragma unroll
  for (int c = 0; c < NCLS; ++c) red[c][tid] = acc[c];
  __syncthreads();

  for (int off = 128; off > 0; off >>= 1) {
    if (tid < off) {
      #pragma unroll
      for (int c = 0; c < NCLS; ++c) red[c][tid] += red[c][tid + off];
    }
    __syncthreads();
  }
  if (tid < NCLS) out[b * NCLS + tid] = red[tid][0] + (float)T_STEPS * b2[tid];
}

extern "C" void kernel_launch(void* const* d_in, const int* in_sizes, int n_in,
                              void* d_out, int out_size, void* d_ws, size_t ws_size,
                              hipStream_t stream) {
  const float* x  = (const float*)d_in[0];   // [256,100,512]
  const float* W1 = (const float*)d_in[1];   // [1024,512]
  const float* b1 = (const float*)d_in[2];   // [1024]
  const float* W2 = (const float*)d_in[3];   // [8,1024]
  const float* b2 = (const float*)d_in[4];   // [8]

  const size_t XB_ELEMS = (size_t)BATCH * T_STEPS * IN_DIM;   // 13,107,200
  unsigned short* xb = (unsigned short*)d_ws;                 // 26.2 MB bf16 copy of x
  float* spk = (float*)((char*)d_ws + XB_ELEMS * sizeof(unsigned short)); // 1 MB

  x_to_bf16_kernel<<<(unsigned)(XB_ELEMS / 2048), 256, 0, stream>>>(x, xb);
  snn_fused_kernel<<<dim3(HID / HN, BATCH / BM), 128, 0, stream>>>(xb, W1, b1, spk);
  snn_head_kernel<<<BATCH, 256, 0, stream>>>(spk, W2, b2, (float*)d_out);
}
